// ModelGraphPolicy_74294344286850
// MI455X (gfx1250) — matrix operand order repaired and checked by reading, not compiled, not run on hardware
//
#include <hip/hip_runtime.h>

typedef float    v2f  __attribute__((ext_vector_type(2)));
typedef float    v4f  __attribute__((ext_vector_type(4)));
typedef float    v8f  __attribute__((ext_vector_type(8)));
typedef _Float16 v2h  __attribute__((ext_vector_type(2)));
typedef _Float16 v8h  __attribute__((ext_vector_type(8)));
typedef _Float16 v16h __attribute__((ext_vector_type(16)));

#define NWAVES 8
#define TB 16

// ---- workspace layout (bytes) ----
#define W2H_WS 0                 // f16 300 x 416 padded dyn_w2  (249600 B)
#define W1H_WS 249600            // f16 400 x 64  dyn_w1[:, :64] (51200 B)
#define WS_NEEDED 300800

// ---- LDS byte offsets (all 16B aligned) ----
#define XU_OFF   0      // f32 16 x 17             (1088 B)
#define A1_OFF   1088   // f32 16 x 65             (4160 B)  z1/a1, reused z3; col64 = inv
#define A2_OFF   5248   // f32 16 x 65             (4160 B)
#define D0H_OFF  9408   // f16 16 x 72  (K=64 pad)  (2304 B)
#define HA_OFF   11712  // f16 16 x 424 (K=416)     (13568 B)
#define HB_OFF   25280  // f16 16 x 328 (K=320)     (10496 B)
#define W3H_OFF  35776  // f16 32 x 320 padded dyn_w3 (20480 B)
#define SMEM_BYTES 56256

__device__ __forceinline__ v8f wmma4(v2f a, v2f b, v8f c) {
  return __builtin_amdgcn_wmma_f32_16x16x4_f32(false, a, false, b, (short)0, c,
                                               false, false);
}
__device__ __forceinline__ v8f wmma16(v16h a, v16h b, v8f c) {
  return __builtin_amdgcn_wmma_f32_16x16x32_f16(false, a, false, b, (short)0, c,
                                                false, false);
}

// Load 2x8 consecutive fp32, convert to f16 A-operand chunk pair (fallback).
template <bool AL>
__device__ __forceinline__ v16h load_cvt8x2(const float* __restrict__ p0,
                                            const float* __restrict__ p1) {
  v16h a;
  if (AL) {
    v4f x0 = *(const v4f*)p0, x1 = *(const v4f*)(p0 + 4);
    v4f y0 = *(const v4f*)p1, y1 = *(const v4f*)(p1 + 4);
#pragma unroll
    for (int i = 0; i < 4; ++i) {
      a[i] = (_Float16)x0[i];      a[4 + i]  = (_Float16)x1[i];
      a[8 + i] = (_Float16)y0[i];  a[12 + i] = (_Float16)y1[i];
    }
  } else {
#pragma unroll
    for (int i = 0; i < 8; ++i) {
      a[i] = (_Float16)p0[i];
      a[8 + i] = (_Float16)p1[i];
    }
  }
  return a;
}

// B operand: 16 contiguous f16 (two 16B loads)
__device__ __forceinline__ v16h ld_b16(const _Float16* __restrict__ p) {
  v8h b0 = *(const v8h*)p;
  v8h b1 = *(const v8h*)(p + 8);
  return __builtin_shufflevector(b0, b1, 0, 1, 2, 3, 4, 5, 6, 7, 8, 9, 10, 11,
                                 12, 13, 14, 15);
}

// A operand from f16 row: chunks [k0+8hf, +8) and [k0+16+8hf, +8)
__device__ __forceinline__ v16h ld_a16(const _Float16* __restrict__ wr, int k0,
                                       int hf) {
  v8h a0 = *(const v8h*)(wr + k0 + 8 * hf);
  v8h a1 = *(const v8h*)(wr + k0 + 16 + 8 * hf);
  return __builtin_shufflevector(a0, a1, 0, 1, 2, 3, 4, 5, 6, 7, 8, 9, 10, 11,
                                 12, 13, 14, 15);
}

// ---------- fp32 WMMA layer (up-net): dst[n][feat] = act(W.src + b) ----------
template <int ACT>  // 0=none, 1=tanh
__device__ __forceinline__ void mlp_layer32(
    const float* __restrict__ W, const float* __restrict__ bias, int wstride,
    int ntiles, int ksteps, int kmax, const float* __restrict__ src,
    int sstride, float* __restrict__ dst, int dstride, int tid) {
  const int n = tid & 15;
  const int hf = (tid >> 4) & 1;
  const int wave = tid >> 5;
  const float* sp = src + n * sstride + 2 * hf;
  for (int mt = wave; mt < ntiles; mt += NWAVES) {
    const float* wr = W + (long)(16 * mt + n) * wstride;
    v8f acc = {0.f, 0.f, 0.f, 0.f, 0.f, 0.f, 0.f, 0.f};
    for (int ks = 0; ks < ksteps; ++ks) {
      int c0 = 4 * ks + 2 * hf;
      int cc0 = (c0 < kmax) ? c0 : kmax;          // K-pad clamp (acts are 0)
      int cc1 = (c0 + 1 < kmax) ? c0 + 1 : kmax;
      v2f a; a.x = wr[cc0]; a.y = wr[cc1];
      v2f b; b.x = sp[4 * ks]; b.y = sp[4 * ks + 1];
      acc = wmma4(a, b, acc);
    }
#pragma unroll
    for (int r = 0; r < 8; ++r) {
      int feat = 16 * mt + 8 * hf + r;
      float v = acc[r] + bias[feat];
      if (ACT == 1) v = tanhf(v);
      dst[n * dstride + feat] = v;
    }
  }
}

// ---- f16 WMMA layer; A from pre-converted f16 (PRE) or fp32+cvt fallback ----
template <bool PRE, bool AL>
__device__ __forceinline__ void mlp_layer16(
    const float* __restrict__ W, const _Float16* __restrict__ Wh,
    const float* __restrict__ bias, int wstride, int whstride, int ntiles,
    int outreal, int ksteps, int maxbase, const _Float16* __restrict__ src,
    int sstride, _Float16* __restrict__ dst, int dstride, int tid) {
  const int n = tid & 15;
  const int hf = (tid >> 4) & 1;
  const int wave = tid >> 5;
  const _Float16* sp = src + n * sstride;
  for (int mt = wave; mt < ntiles; mt += 2 * NWAVES) {
    const int mtb = mt + NWAVES;
    const bool has2 = (mtb < ntiles);
    const int mtbc = has2 ? mtb : mt;
    int rowa = 16 * mt + n;   if (rowa >= outreal) rowa = outreal - 1;
    int rowb = 16 * mtbc + n; if (rowb >= outreal) rowb = outreal - 1;
    const float* wa = PRE ? nullptr : W + (long)rowa * wstride;
    const float* wb = PRE ? nullptr : W + (long)rowb * wstride;
    const _Float16* wah = PRE ? Wh + (long)rowa * whstride : nullptr;
    const _Float16* wbh = PRE ? Wh + (long)rowb * whstride : nullptr;
    v8f acc0 = {0.f, 0.f, 0.f, 0.f, 0.f, 0.f, 0.f, 0.f};
    v8f acc1 = {0.f, 0.f, 0.f, 0.f, 0.f, 0.f, 0.f, 0.f};
    for (int ks = 0; ks < ksteps; ++ks) {
      int k0 = 32 * ks;
      v16h b = ld_a16(sp, k0, hf);            // two ds_load_b128
      v16h a0, a1;
      if (PRE) {
        a0 = ld_a16(wah, k0, hf);             // two global_load_b128 (f16)
        a1 = ld_a16(wbh, k0, hf);
      } else {
        int cb0 = k0 + 8 * hf;      if (cb0 > maxbase) cb0 = maxbase;
        int cb1 = k0 + 16 + 8 * hf; if (cb1 > maxbase) cb1 = maxbase;
        a0 = load_cvt8x2<AL>(wa + cb0, wa + cb1);
        a1 = load_cvt8x2<AL>(wb + cb0, wb + cb1);
      }
      acc0 = wmma16(a0, b, acc0);
      acc1 = wmma16(a1, b, acc1);
    }
#pragma unroll
    for (int r = 0; r < 8; r += 2) {   // relu + bias, pack pairs -> b32 store
      int feat = 16 * mt + 8 * hf + r;
      v2h p; p.x = (_Float16)0.f; p.y = (_Float16)0.f;
      if (feat < outreal) {
        p.x = (_Float16)fmaxf(acc0[r] + bias[feat], 0.f);
        p.y = (_Float16)fmaxf(acc0[r + 1] + bias[feat + 1], 0.f);
      }
      *(v2h*)(dst + (long)n * dstride + feat) = p;
    }
    if (has2) {
#pragma unroll
      for (int r = 0; r < 8; r += 2) {
        int feat = 16 * mtb + 8 * hf + r;
        v2h p; p.x = (_Float16)0.f; p.y = (_Float16)0.f;
        if (feat < outreal) {
          p.x = (_Float16)fmaxf(acc1[r] + bias[feat], 0.f);
          p.y = (_Float16)fmaxf(acc1[r + 1] + bias[feat + 1], 0.f);
        }
        *(v2h*)(dst + (long)n * dstride + feat) = p;
      }
    }
  }
}

// ---------- D3: f16 WMMA, A (padded dyn_w3) and B both from LDS ----------
__device__ __forceinline__ void mlp_layer16_d3(
    const _Float16* __restrict__ Wl, const float* __restrict__ bias,
    const _Float16* __restrict__ src, int sstride, float* __restrict__ out,
    int tid) {
  const int n = tid & 15;
  const int hf = (tid >> 4) & 1;
  const int wave = tid >> 5;
  for (int mt = wave; mt < 2; mt += NWAVES) {
    const _Float16* wr = Wl + (16 * mt + n) * 320;
    v8f acc = {0.f, 0.f, 0.f, 0.f, 0.f, 0.f, 0.f, 0.f};
    for (int ks = 0; ks < 10; ++ks) {
      int k0 = 32 * ks;
      v16h a = ld_a16(wr, k0, hf);
      v16h b = ld_a16(src + n * sstride, k0, hf);
      acc = wmma16(a, b, acc);
    }
#pragma unroll
    for (int r = 0; r < 8; ++r) {
      int feat = 16 * mt + 8 * hf + r;
      out[(long)n * 32 + feat] = acc[r] + bias[feat];
    }
  }
}

// ---- one-time weight pre-conversion (fp32 -> padded f16 in d_ws) ----
__global__ void __launch_bounds__(256) cvt_weights_kernel(
    const float* __restrict__ dyn_w1, const float* __restrict__ dyn_w2,
    _Float16* __restrict__ w1h, _Float16* __restrict__ w2h) {
  int i = blockIdx.x * 256 + threadIdx.x;
  if (i < 300 * 416) {
    int m = i / 416, k = i - m * 416;
    w2h[i] = (k < 400) ? (_Float16)dyn_w2[m * 400 + k] : (_Float16)0.f;
  }
  if (i < 400 * 64) {
    int m = i >> 6, k = i & 63;
    w1h[i] = (_Float16)dyn_w1[m * 65 + k];
  }
}

__global__ void __launch_bounds__(256) mgp_fused_kernel(
    const float* __restrict__ state, const float* __restrict__ action,
    const float* __restrict__ up_w1, const float* __restrict__ up_b1,
    const float* __restrict__ up_w2, const float* __restrict__ up_b2,
    const float* __restrict__ up_w3, const float* __restrict__ up_b3,
    const float* __restrict__ dyn_w1, const float* __restrict__ dyn_b1,
    const float* __restrict__ dyn_w2, const float* __restrict__ dyn_b2,
    const float* __restrict__ dyn_w3, const float* __restrict__ dyn_b3,
    const _Float16* __restrict__ w1h, const _Float16* __restrict__ w2h,
    int use_pre, float* __restrict__ out) {
  __shared__ __align__(16) char smem[SMEM_BYTES];
  float* xu = (float*)(smem + XU_OFF);
  float* a1 = (float*)(smem + A1_OFF);
  float* a2 = (float*)(smem + A2_OFF);
  _Float16* d0h = (_Float16*)(smem + D0H_OFF);
  _Float16* hA = (_Float16*)(smem + HA_OFF);
  _Float16* hB = (_Float16*)(smem + HB_OFF);
  _Float16* w3h = (_Float16*)(smem + W3H_OFF);

  const int tid = threadIdx.x;
  const long gb = (long)blockIdx.x * TB;

  {  // stage xu = [state(12), u(1), 0 pad] as f32 [n][k]
    int n = tid >> 4, k = tid & 15;
    float v = 0.f;
    if (k < 12) v = state[(gb + n) * 12 + k];
    else if (k == 12) v = action[gb + n];
    xu[n * 17 + k] = v;
  }
  // stage dyn_w3 -> LDS f16 padded 32x320 (read once per WG, L2-resident)
  for (int i = tid; i < 32 * 320; i += 256) {
    int m = i / 320, k = i - m * 320;
    w3h[i] = (k < 300) ? (_Float16)dyn_w3[m * 300 + k] : (_Float16)0.f;
  }
  // zero K-pad columns of hA (400..415) and hB (304..319)
  {
    int n = tid >> 4, c = tid & 15;
    hA[n * 424 + 400 + c] = (_Float16)0.f;
    hB[n * 328 + 304 + c] = (_Float16)0.f;
  }
  __syncthreads();

  // L1 (fp32): z1 = xu @ up_w1.T + b1              (64, K=13 padded to 16)
  mlp_layer32<0>(up_w1, up_b1, 13, 4, 4, 12, xu, 17, a1, 65, tid);
  __syncthreads();
  // a1 = tanh(l2norm(z1)): 16 threads reduce, 256 threads apply
  if (tid < TB) {
    const float* rp = a1 + tid * 65;
    float s = 0.f;
    for (int f = 0; f < 64; ++f) { float x = rp[f]; s += x * x; }
    a1[tid * 65 + 64] = 1.f / fmaxf(sqrtf(s), 1e-12f);
  }
  __syncthreads();
  {
    int row = tid >> 4, f0 = (tid & 15) * 4;
    float* rp = a1 + row * 65;
    float inv = rp[64];
#pragma unroll
    for (int f = f0; f < f0 + 4; ++f) rp[f] = tanhf(rp[f] * inv);
  }
  __syncthreads();
  // L2 (fp32): a2 = tanh(a1 @ up_w2[:, :64].T + b2) (64, K=64)
  mlp_layer32<1>(up_w2, up_b2, 128, 4, 16, 1 << 28, a1, 65, a2, 65, tid);
  __syncthreads();
  // L3 (fp32): z3 = a2 @ up_w3.T + b3 -> a1 cols 0..31
  mlp_layer32<0>(up_w3, up_b3, 64, 2, 16, 1 << 28, a2, 65, a1, 65, tid);
  __syncthreads();
  // d0h = f16[ l2norm(z3), u, zeros(31) ]  (K padded to 64)
  if (tid < TB) {
    const float* zp = a1 + tid * 65;
    float s = 0.f;
    for (int f = 0; f < 32; ++f) { float x = zp[f]; s += x * x; }
    a1[tid * 65 + 64] = 1.f / fmaxf(sqrtf(s), 1e-12f);
  }
  __syncthreads();
  {
    int row = tid >> 4, seg = tid & 15;
    const float* zp = a1 + row * 65;
    _Float16* dp = d0h + row * 72;
    float inv = zp[64];
    dp[2 * seg]     = (_Float16)(zp[2 * seg] * inv);
    dp[2 * seg + 1] = (_Float16)(zp[2 * seg + 1] * inv);
    if (seg == 0) {
      dp[32] = (_Float16)action[gb + row];
      dp[33] = (_Float16)0.f;
    } else {
      dp[32 + 2 * seg] = (_Float16)0.f;
      dp[33 + 2 * seg] = (_Float16)0.f;
    }
  }
  __syncthreads();
  // D1 (f16): hA = relu([msg,u,0] @ dyn_w1[:, :64].T + db1)  (400, K=64)
  if (use_pre)
    mlp_layer16<true, false>(nullptr, w1h, dyn_b1, 0, 64, 25, 400, 2, 0, d0h,
                             72, hA, 424, tid);
  else
    mlp_layer16<false, false>(dyn_w1, nullptr, dyn_b1, 65, 0, 25, 400, 2,
                              1 << 28, d0h, 72, hA, 424, tid);
  __syncthreads();
  // D2 (f16): hB = relu(hA @ dyn_w2.T + db2)       (300 pad 304, K=416)
  if (use_pre)
    mlp_layer16<true, true>(nullptr, w2h, dyn_b2, 0, 416, 19, 300, 13, 0, hA,
                            424, hB, 328, tid);
  else
    mlp_layer16<false, true>(dyn_w2, nullptr, dyn_b2, 400, 0, 19, 300, 13, 392,
                             hA, 424, hB, 328, tid);
  __syncthreads();
  // D3 (f16): delta = hB @ dyn_w3.T + db3 -> global (32, K=320)
  mlp_layer16_d3(w3h, dyn_b3, hB, 328, out + gb * 32, tid);
}

extern "C" void kernel_launch(void* const* d_in, const int* in_sizes, int n_in,
                              void* d_out, int out_size, void* d_ws,
                              size_t ws_size, hipStream_t stream) {
  const float* state = (const float*)d_in[0];
  const float* action = (const float*)d_in[1];
  const float* up_w1 = (const float*)d_in[2];
  const float* up_b1 = (const float*)d_in[3];
  const float* up_w2 = (const float*)d_in[4];
  const float* up_b2 = (const float*)d_in[5];
  const float* up_w3 = (const float*)d_in[6];
  const float* up_b3 = (const float*)d_in[7];
  const float* dyn_w1 = (const float*)d_in[8];
  const float* dyn_b1 = (const float*)d_in[9];
  const float* dyn_w2 = (const float*)d_in[10];
  const float* dyn_b2 = (const float*)d_in[11];
  const float* dyn_w3 = (const float*)d_in[12];
  const float* dyn_b3 = (const float*)d_in[13];
  float* out = (float*)d_out;

  const int use_pre = (d_ws != nullptr && ws_size >= (size_t)WS_NEEDED) ? 1 : 0;
  _Float16* w2h = (_Float16*)((char*)d_ws + W2H_WS);
  _Float16* w1h = (_Float16*)((char*)d_ws + W1H_WS);
  if (use_pre) {
    const int total = 300 * 416;  // dominant conversion count
    cvt_weights_kernel<<<(total + 255) / 256, 256, 0, stream>>>(dyn_w1, dyn_w2,
                                                               w1h, w2h);
  }

  const int B = in_sizes[1];  // action is (B,1)
  const int blocks = B / TB;  // 262144/16 = 16384
  mgp_fused_kernel<<<blocks, 256, 0, stream>>>(
      state, action, up_w1, up_b1, up_w2, up_b2, up_w3, up_b3, dyn_w1, dyn_b1,
      dyn_w2, dyn_b2, dyn_w3, dyn_b3, w1h, w2h, use_pre, out);
}
